// InteractionLayer_80006650790187
// MI455X (gfx1250) — compile-verified, matrix operand order checked by reading, not run on hardware
//
#include <hip/hip_runtime.h>
#include <hip/hip_bf16.h>
#include <math.h>

// ---------------------------------------------------------------------------
// Problem constants (from reference)
// ---------------------------------------------------------------------------
constexpr int kNS   = 20000;
constexpr int kNT   = 20000;
constexpr long kE   = 320000;
constexpr int kH    = 128;      // hidden
constexpr int kSPHE = 12;       // 3*2^2
constexpr int kTORS = 6;        // 3*2
constexpr float kCUT = 5.0f;
constexpr float kPI  = 3.14159265358979323846f;

typedef __bf16 bf16_t;
typedef unsigned short u16;
typedef __attribute__((ext_vector_type(16))) __bf16 v16bf;
typedef __attribute__((ext_vector_type(8)))  float  v8f;

// ---------------------------------------------------------------------------
// Helpers
// ---------------------------------------------------------------------------
__device__ __forceinline__ u16 f2bfu(float f) {
    unsigned u = __builtin_bit_cast(unsigned, f);
    return (u16)((u + 0x7FFFu + ((u >> 16) & 1u)) >> 16);   // RNE
}
__device__ __forceinline__ bf16_t f2bf(float f) {
    u16 s = f2bfu(f);
    return __builtin_bit_cast(bf16_t, s);
}
__device__ __forceinline__ float bf2f(bf16_t b) {
    unsigned u = ((unsigned)__builtin_bit_cast(u16, b)) << 16;
    return __builtin_bit_cast(float, u);
}
__device__ __forceinline__ float silu_f(float x) {
    return x / (1.0f + __expf(-x));
}

// K-offset inside a 32-wide K window for A/B register r (0..7) per ISA layout:
// lanes 0-15: regs 0-3 -> K 0..7 (pairs), regs 4-7 -> K 16..23; lanes 16-31: +8
__device__ __forceinline__ int kmap(int r, int hi) {
    return (r < 4 ? 2 * r : 8 + 2 * r) + (hi ? 8 : 0);
}

// Element (m, k) of a 16xK A-tile -> index into fragment-layout buffer
// (fragment layout: [k/32 chunk][lane][16 bf16], lane fragment is contiguous)
__device__ __forceinline__ int a_frag_idx(int m, int k) {
    int kc  = k >> 5, kin = k & 31;
    int hi  = (kin >> 3) & 1;               // which 16-lane half holds this k
    int lane = m + (hi << 4);
    int kp  = kin - (hi << 3);              // kp in {0..7} U {16..23}
    int r   = (kp < 8) ? (kp >> 1) : (4 + ((kp - 16) >> 1));
    return ((kc << 5) + lane) * 16 + ((r << 1) + (kin & 1));
}
// Same for a 32xK A-buffer (two stacked 16-row fragment regions).
__device__ __forceinline__ int fidx32(int m, int k, int K) {
    return (m >> 4) * 16 * K + a_frag_idx(m & 15, k);
}

__device__ __forceinline__ v8f wmma_bf16(v16bf a, v16bf b, v8f c) {
    return __builtin_amdgcn_wmma_f32_16x16x32_bf16(false, a, false, b, (short)0, c,
                                                   false, false);
}

// 16 rows: C = A(16xK frags, LDS) * B(KxN bf16 frags, global) at tile nt.
__device__ __forceinline__ v8f gemm_tile(const bf16_t* __restrict__ Afrag, int K,
                                         const u16* __restrict__ Wfrag, int nt,
                                         int lane) {
    const v16bf* A = (const v16bf*)Afrag;
    const v16bf* B = (const v16bf*)Wfrag + (size_t)nt * (K >> 5) * 32;
    v8f c = {};
#pragma unroll
    for (int kc = 0; kc < K / 32; ++kc) {
        v16bf a = A[kc * 32 + lane];
        v16bf b = B[kc * 32 + lane];
        c = wmma_bf16(a, b, c);
    }
    return c;
}

// 32 rows: each B fragment loaded ONCE, reused for two WMMAs (rows 0-15, 16-31).
__device__ __forceinline__ void gemm_x2(const bf16_t* __restrict__ Afrag, int K,
                                        const u16* __restrict__ Wfrag, int nt,
                                        int lane, v8f& c0, v8f& c1) {
    const v16bf* A0 = (const v16bf*)Afrag;
    const v16bf* A1 = A0 + K;                       // second 16-row region
    const v16bf* B  = (const v16bf*)Wfrag + (size_t)nt * (K >> 5) * 32;
    c0 = {}; c1 = {};
#pragma unroll
    for (int kc = 0; kc < K / 32; ++kc) {
        v16bf b  = B[kc * 32 + lane];
        c0 = wmma_bf16(A0[kc * 32 + lane], b, c0);
        c1 = wmma_bf16(A1[kc * 32 + lane], b, c1);
    }
}

// Store a C/D tile to row-major fp32 [16][ldn] at column n0.
__device__ __forceinline__ void store16(float* D, int ldn, int n0, int lane, v8f c) {
    int n = n0 + (lane & 15), mb = (lane >> 4) * 8;
#pragma unroll
    for (int r = 0; r < 8; ++r) D[(mb + r) * ldn + n] = c[r];
}

// ---------------------------------------------------------------------------
// Kernel P: one-time weight pre-swizzle fp32 (KxN row-major) -> bf16 B-fragments
// ---------------------------------------------------------------------------
__global__ void prep_weight_kernel(const float* __restrict__ W, int K, int N,
                                   u16* __restrict__ out) {
    int total = K * N;
    int kcs = K >> 5;
    for (int t = blockIdx.x * blockDim.x + threadIdx.x; t < total;
         t += gridDim.x * blockDim.x) {
        int pos = t & 15;
        int frag = t >> 4;
        int lane = frag & 31;
        int tk = frag >> 5;            // nt * kcs + kc
        int kc = tk % kcs;
        int nt = tk / kcs;
        int r = pos >> 1;
        int hi = lane >> 4;
        int k = (kc << 5) + kmap(r, hi) + (pos & 1);
        int n = (nt << 4) + (lane & 15);
        out[t] = f2bfu(W[(long)k * N + n]);
    }
}

// ---------------------------------------------------------------------------
// Kernel 0: zero segment-sum accumulators
// ---------------------------------------------------------------------------
__global__ void zero_kernel(float* p, long n) {
    long i = (long)blockIdx.x * blockDim.x + threadIdx.x;
    long stride = (long)gridDim.x * blockDim.x;
    for (; i < n; i += stride) p[i] = 0.0f;
}

// ---------------------------------------------------------------------------
// Kernel 1: q = x_t@Wq+bq, k = x_s@Wk+bk, v = x_s@Wv+bv  (16-row tiles, WMMA)
// ---------------------------------------------------------------------------
__global__ void __launch_bounds__(256)
qkv_kernel(const float* __restrict__ x_s, const float* __restrict__ x_t,
           const u16* __restrict__ fWq, const float* __restrict__ bq,
           const u16* __restrict__ fWk, const float* __restrict__ bk,
           const u16* __restrict__ fWv, const float* __restrict__ bv,
           float* __restrict__ qbuf, float* __restrict__ kbuf, float* __restrict__ vbuf) {
    __shared__ __align__(32) bf16_t xt_bf[16 * kH];
    __shared__ __align__(32) bf16_t xs_bf[16 * kH];
    const int tid = threadIdx.x, lane = tid & 31, wave = tid >> 5;
    const long r0 = (long)blockIdx.x * 16;

    for (int i = tid; i < 16 * kH; i += 256) {
        int m = i >> 7, k = i & 127;
        int fi = a_frag_idx(m, k);
        xt_bf[fi] = f2bf(x_t[r0 * kH + i]);
        xs_bf[fi] = f2bf(x_s[r0 * kH + i]);
    }
    __syncthreads();

    for (int t = wave; t < 24; t += 8) {        // 3 matrices x 8 N-tiles
        int mat = t >> 3, nt = t & 7;
        const bf16_t* A = (mat == 0) ? xt_bf : xs_bf;
        const u16* W    = (mat == 0) ? fWq : (mat == 1 ? fWk : fWv);
        const float* B  = (mat == 0) ? bq : (mat == 1 ? bk : bv);
        float* O        = (mat == 0) ? qbuf : (mat == 1 ? kbuf : vbuf);
        v8f c = gemm_tile(A, kH, W, nt, lane);
        int n = nt * 16 + (lane & 15), mb = (lane >> 4) * 8;
#pragma unroll
        for (int r = 0; r < 8; ++r)
            O[(r0 + mb + r) * kH + n] = c[r] + B[n];
    }
}

// ---------------------------------------------------------------------------
// Kernel 2: fused edge pipeline, 32 edges per block (B-fragment reuse x2)
// ---------------------------------------------------------------------------
__global__ void __launch_bounds__(256)
edge_kernel(const float* __restrict__ edge_attr,
            const int* __restrict__ src_idx, const int* __restrict__ tgt_idx,
            const float* __restrict__ sphe, const float* __restrict__ tors,
            const float* __restrict__ dist,
            const float* __restrict__ Ws1, const u16* __restrict__ fWs2,
            const float* __restrict__ Wt1, const u16* __restrict__ fWt2,
            const u16* __restrict__ fWdk, const float* __restrict__ bdk,
            const u16* __restrict__ fWdv, const float* __restrict__ bdv,
            const u16* __restrict__ fWo_att, const float* __restrict__ bo_att,
            const u16* __restrict__ fWs_msg, const float* __restrict__ bs_msg,
            const u16* __restrict__ fWf, const float* __restrict__ bfv,
            const u16* __restrict__ fWcat_f, const float* __restrict__ bcat_f,
            const float* __restrict__ qbuf, const float* __restrict__ kbuf,
            const float* __restrict__ vbuf,
            float* __restrict__ s_msg_acc, float* __restrict__ dsum_acc,
            float* __restrict__ fji_out) {
    // Manually-unioned ~61 KB LDS arena; M = 32 edges.
    __shared__ __align__(32) char smem[61056];
    bf16_t* ea_bf  = (bf16_t*)(smem);           // 32x128 A-frags            (8K)
    bf16_t* d1b    = (bf16_t*)(smem + 8192);    // 32x128 bf16 row-major d1  (8K)
    bf16_t* d2b    = (bf16_t*)(smem + 16384);   // 32x128 bf16 row-major d2  (8K)
    bf16_t* slotA  = (bf16_t*)(smem + 24576);   // dk -> later f1 (row-major)(8K)
    bf16_t* slotB  = (bf16_t*)(smem + 32768);   // dv (row-major)            (8K)
    bf16_t* slotCD = (bf16_t*)(smem + 40960);   // mid|vjpre|vj -> g frags  (16K)
    float*  sphe_s = (float*)(smem + 57344);    // 32x12
    float*  tors_s = (float*)(smem + 58880);    // 32x6
    float*  dist_s = (float*)(smem + 59648);    // 32
    int*    src_s  = (int*)(smem + 59776);      // 32
    int*    tgt_s  = (int*)(smem + 59904);      // 32
    float*  attn_s = (float*)(smem + 60032);    // 32x8

    const int tid = threadIdx.x, lane = tid & 31, wave = tid >> 5;
    const long e0 = (long)blockIdx.x * 32;
    const int lo = lane & 15, mb = (lane >> 4) * 8;

    // ---- Stage 0: stage edge tile ----
    for (int i = tid; i < 32 * kH; i += 256) {
        int m = i >> 7, k = i & 127;
        ea_bf[fidx32(m, k, kH)] = f2bf(edge_attr[e0 * kH + i]);
    }
    for (int i = tid; i < 32 * kSPHE; i += 256) sphe_s[i] = sphe[e0 * kSPHE + i];
    for (int i = tid; i < 32 * kTORS; i += 256) tors_s[i] = tors[e0 * kTORS + i];
    if (tid < 32) {
        dist_s[tid] = dist[e0 + tid];
        src_s[tid]  = src_idx[e0 + tid];
        tgt_s[tid]  = tgt_idx[e0 + tid];
    }
    __syncthreads();

    // ---- Stage 1: mid1 = sphe@Ws1, mid2 = tors@Wt1 (tiny K, scalar f32) ----
    bf16_t* mid1 = slotCD;              // 32x64 frags (4K)
    bf16_t* mid2 = slotCD + 32 * 64;    // 32x64 frags (4K)
    for (int idx = tid; idx < 32 * 64; idx += 256) {
        int m = idx >> 6, j = idx & 63;
        float s1 = 0.f, s2 = 0.f;
#pragma unroll
        for (int i = 0; i < kSPHE; ++i) s1 += sphe_s[m * kSPHE + i] * Ws1[i * 64 + j];
#pragma unroll
        for (int i = 0; i < kTORS; ++i) s2 += tors_s[m * kTORS + i] * Wt1[i * 64 + j];
        int fi = fidx32(m, j, 64);
        mid1[fi] = f2bf(s1);
        mid2[fi] = f2bf(s2);
    }
    __syncthreads();

    // ---- Stage 2: d1 = mid1@Ws2, d2 = mid2@Wt2 (WMMA, K=64) ----
    for (int t = wave; t < 16; t += 8) {
        const bf16_t* A = (t < 8) ? mid1 : mid2;
        const u16*    W = (t < 8) ? fWs2 : fWt2;
        bf16_t* D      = (t < 8) ? d1b : d2b;
        v8f c0, c1;
        gemm_x2(A, 64, W, t & 7, lane, c0, c1);
        int n = (t & 7) * 16 + lo;
#pragma unroll
        for (int r = 0; r < 8; ++r) {
            D[(mb + r) * kH + n]      = f2bf(c0[r]);
            D[(16 + mb + r) * kH + n] = f2bf(c1[r]);
        }
    }
    __syncthreads();

    // ---- Stage 3: dk = silu(ea@Wdk+b), dv = silu(ea@Wdv+b) (WMMA) ----
    {
        v8f a0, a1, b0, b1;
        gemm_x2(ea_bf, kH, fWdk, wave, lane, a0, a1);
        gemm_x2(ea_bf, kH, fWdv, wave, lane, b0, b1);
        int n = wave * 16 + lo;
        float bk_ = bdk[n], bv_ = bdv[n];
#pragma unroll
        for (int r = 0; r < 8; ++r) {
            slotA[(mb + r) * kH + n]      = f2bf(silu_f(a0[r] + bk_));
            slotA[(16 + mb + r) * kH + n] = f2bf(silu_f(a1[r] + bk_));
            slotB[(mb + r) * kH + n]      = f2bf(silu_f(b0[r] + bv_));
            slotB[(16 + mb + r) * kH + n] = f2bf(silu_f(b1[r] + bv_));
        }
    }
    __syncthreads();

    // ---- Stage 4: attention logits per (edge, head): 32*8 = 256 threads ----
    {
        int m = tid >> 3, h = tid & 7;
        const float4* q4 = (const float4*)(qbuf + (long)tgt_s[m] * kH + h * 16);
        const float4* k4 = (const float4*)(kbuf + (long)src_s[m] * kH + h * 16);
        const bf16_t* dkr = slotA + m * kH + h * 16;
        float s = 0.f;
#pragma unroll
        for (int jj = 0; jj < 4; ++jj) {
            float4 qv = q4[jj], kv = k4[jj];
            s += qv.x * kv.x * bf2f(dkr[jj * 4 + 0]);
            s += qv.y * kv.y * bf2f(dkr[jj * 4 + 1]);
            s += qv.z * kv.z * bf2f(dkr[jj * 4 + 2]);
            s += qv.w * kv.w * bf2f(dkr[jj * 4 + 3]);
        }
        float d = dist_s[m];
        float cut = (d < kCUT) ? 0.5f * (__cosf(kPI * d / kCUT) + 1.0f) : 0.0f;
        attn_s[m * 8 + h] = silu_f(s) * cut;
    }
    __syncthreads();

    // ---- Stage 5: vjpre = v[src] * dv * attn (A-frags in lower slotCD) ----
    {
        bf16_t* vjpre = slotCD;          // mid1/mid2 dead
        for (int idx = tid; idx < 32 * kH; idx += 256) {
            int m = idx >> 7, n = idx & 127;
            float val = vbuf[(long)src_s[m] * kH + n] * bf2f(slotB[idx]) *
                        attn_s[m * 8 + (n >> 4)];
            vjpre[fidx32(m, n, kH)] = f2bf(val);
        }
    }
    __syncthreads();

    // ---- Stage 6: v_j = silu(vjpre@Wo_att+b); accumulate s_msg ----
    bf16_t* vjb = slotCD + 32 * kH;      // upper half of slotCD (8K)
    {
        v8f c0, c1;
        gemm_x2(slotCD, kH, fWo_att, wave, lane, c0, c1);
        int n = wave * 16 + lo;
        float bo_ = bo_att[n];
#pragma unroll
        for (int r = 0; r < 8; ++r) {
            float y0 = silu_f(c0[r] + bo_), y1 = silu_f(c1[r] + bo_);
            vjb[fidx32(mb + r, n, kH)]      = f2bf(y0);
            vjb[fidx32(16 + mb + r, n, kH)] = f2bf(y1);
            atomicAdd(&s_msg_acc[(long)tgt_s[mb + r] * kH + n], y0);
            atomicAdd(&s_msg_acc[(long)tgt_s[16 + mb + r] * kH + n], y1);
        }
    }
    __syncthreads();

    // ---- Stage 7: s = silu(vj@Ws_msg+b); d_ji = cat(s1*d1, s2*d2); acc dsum ----
    for (int t = wave; t < 16; t += 8) {
        v8f c0, c1;
        gemm_x2(vjb, kH, fWs_msg, t, lane, c0, c1);
        int n = t * 16 + lo;
        float bs_ = bs_msg[n];
        const bf16_t* dsrc = (n < kH) ? d1b : d2b;
        int nd = (n < kH) ? n : n - kH;
#pragma unroll
        for (int r = 0; r < 8; ++r) {
            float y0 = silu_f(c0[r] + bs_) * bf2f(dsrc[(mb + r) * kH + nd]);
            float y1 = silu_f(c1[r] + bs_) * bf2f(dsrc[(16 + mb + r) * kH + nd]);
            atomicAdd(&dsum_acc[(long)tgt_s[mb + r] * 2 * kH + n], y0);
            atomicAdd(&dsum_acc[(long)tgt_s[16 + mb + r] * 2 * kH + n], y1);
        }
    }
    __syncthreads();                      // vj dead; slotCD becomes g

    // ---- Stage 8: f = silu(ea@Wf+b); f1 -> slotA, g = cat(f2*d1, f3*d2) ----
    {
        bf16_t* f1b = slotA;              // dk dead after stage 4
        bf16_t* g   = slotCD;             // 32x256 frags (16K)
        for (int t = wave; t < 24; t += 8) {
            v8f c0, c1;
            gemm_x2(ea_bf, kH, fWf, t, lane, c0, c1);
            int n = t * 16 + lo;
            float bf_ = bfv[n];
#pragma unroll
            for (int r = 0; r < 8; ++r) {
                float y0 = silu_f(c0[r] + bf_), y1 = silu_f(c1[r] + bf_);
                if (n < kH) {
                    f1b[(mb + r) * kH + n]      = f2bf(y0);
                    f1b[(16 + mb + r) * kH + n] = f2bf(y1);
                } else {
                    const bf16_t* dsrc = (n < 2 * kH) ? d1b : d2b;
                    int nd = (n < 2 * kH) ? n - kH : n - 2 * kH;
                    g[fidx32(mb + r, n - kH, 2 * kH)] =
                        f2bf(y0 * bf2f(dsrc[(mb + r) * kH + nd]));
                    g[fidx32(16 + mb + r, n - kH, 2 * kH)] =
                        f2bf(y1 * bf2f(dsrc[(16 + mb + r) * kH + nd]));
                }
            }
        }
    }
    __syncthreads();

    // ---- Stage 9: f_ji = f1 + silu(g@Wcat_f + b) -> output ----
    {
        v8f c0, c1;
        gemm_x2(slotCD, 2 * kH, fWcat_f, wave, lane, c0, c1);
        int n = wave * 16 + lo;
        float bc_ = bcat_f[n];
#pragma unroll
        for (int r = 0; r < 8; ++r) {
            fji_out[(e0 + mb + r) * kH + n] =
                bf2f(slotA[(mb + r) * kH + n]) + silu_f(c0[r] + bc_);
            fji_out[(e0 + 16 + mb + r) * kH + n] =
                bf2f(slotA[(16 + mb + r) * kH + n]) + silu_f(c1[r] + bc_);
        }
    }
}

// ---------------------------------------------------------------------------
// Kernel 3: node-level tail: d_msg, o, h_t
// ---------------------------------------------------------------------------
__global__ void __launch_bounds__(256)
final_kernel(const float* __restrict__ x_t,
             const float* __restrict__ s_msg_acc, const float* __restrict__ dsum_acc,
             const u16* __restrict__ fWcat_msg,
             const u16* __restrict__ fWo_msg, const float* __restrict__ bo_msg,
             const u16* __restrict__ fWfinal, const float* __restrict__ bfinal,
             float* __restrict__ ht_out) {
    __shared__ __align__(32) char smem[45 * 1024];
    bf16_t* dsum_bf = (bf16_t*)smem;              // 16x256 frags (8K)
    bf16_t* smsg_bf = (bf16_t*)(smem + 8192);     // 16x128 frags (4K)
    float*  dmsg    = (float*)(smem + 12288);     // 16x128 f32 (8K)
    float*  obuf    = (float*)(smem + 20480);     // 16x256 f32 (16K)
    bf16_t* cat_bf  = (bf16_t*)(smem + 36864);    // 16x256 frags (8K)
    const int tid = threadIdx.x, lane = tid & 31, wave = tid >> 5;
    const long r0 = (long)blockIdx.x * 16;

    for (int i = tid; i < 16 * 2 * kH; i += 256) {
        int m = i >> 8, j = i & 255;
        dsum_bf[a_frag_idx(m, j)] = f2bf(dsum_acc[r0 * 2 * kH + i]);
    }
    for (int i = tid; i < 16 * kH; i += 256) {
        int m = i >> 7, n = i & 127;
        smsg_bf[a_frag_idx(m, n)] = f2bf(s_msg_acc[r0 * kH + i]);
        cat_bf[a_frag_idx(m, n)]  = f2bf(x_t[r0 * kH + i]);   // cat[:, :H] = x_t
    }
    __syncthreads();

    for (int t = wave; t < 24; t += 8) {
        if (t < 8) {   // d_msg = dsum @ Wcat_msg (K=256)
            v8f c = gemm_tile(dsum_bf, 2 * kH, fWcat_msg, t, lane);
            store16(dmsg, kH, t * 16, lane, c);
        } else {       // o = s_msg @ Wo_msg + bo_msg (N=256)
            int nt = t - 8;
            v8f c = gemm_tile(smsg_bf, kH, fWo_msg, nt, lane);
            int n = nt * 16 + (lane & 15), mb = (lane >> 4) * 8;
#pragma unroll
            for (int r = 0; r < 8; ++r) obuf[(mb + r) * 2 * kH + n] = c[r] + bo_msg[n];
        }
    }
    __syncthreads();

    for (int i = tid; i < 16 * kH; i += 256) {   // z = s_msg1 + s_msg2 * d_msg
        int m = i >> 7, n = i & 127;
        float z = obuf[m * 2 * kH + n] + obuf[m * 2 * kH + kH + n] * dmsg[i];
        cat_bf[a_frag_idx(m, kH + n)] = f2bf(z);
    }
    __syncthreads();

    {   // h_t = silu(cat @ Wfinal + bfinal)
        v8f c = gemm_tile(cat_bf, 2 * kH, fWfinal, wave, lane);
        int n = wave * 16 + (lane & 15), mb = (lane >> 4) * 8;
#pragma unroll
        for (int r = 0; r < 8; ++r)
            ht_out[(r0 + mb + r) * kH + n] = silu_f(c[r] + bfinal[n]);
    }
}

// ---------------------------------------------------------------------------
// Launch
// ---------------------------------------------------------------------------
extern "C" void kernel_launch(void* const* d_in, const int* in_sizes, int n_in,
                              void* d_out, int out_size, void* d_ws, size_t ws_size,
                              hipStream_t stream) {
    (void)in_sizes; (void)n_in; (void)out_size; (void)ws_size;
    const float* x_s       = (const float*)d_in[0];
    const float* x_t       = (const float*)d_in[1];
    const int*   ei        = (const int*)d_in[2];
    const float* edge_attr = (const float*)d_in[3];
    const float* sphe      = (const float*)d_in[4];
    const float* tors      = (const float*)d_in[5];
    const float* dist      = (const float*)d_in[6];
    const float* Ws1 = (const float*)d_in[7],  *Ws2 = (const float*)d_in[8];
    const float* Wt1 = (const float*)d_in[9],  *Wt2 = (const float*)d_in[10];
    const float* Wq  = (const float*)d_in[11], *bq  = (const float*)d_in[12];
    const float* Wk  = (const float*)d_in[13], *bk  = (const float*)d_in[14];
    const float* Wv  = (const float*)d_in[15], *bv  = (const float*)d_in[16];
    const float* Wdk = (const float*)d_in[17], *bdk = (const float*)d_in[18];
    const float* Wdv = (const float*)d_in[19], *bdv = (const float*)d_in[20];
    const float* Wo_att  = (const float*)d_in[21], *bo_att  = (const float*)d_in[22];
    const float* Ws_msg  = (const float*)d_in[23], *bs_msg  = (const float*)d_in[24];
    const float* Wcat_msg= (const float*)d_in[25];
    const float* Wo_msg  = (const float*)d_in[26], *bo_msg  = (const float*)d_in[27];
    const float* Wfinal  = (const float*)d_in[28], *bfinal  = (const float*)d_in[29];
    const float* Wf      = (const float*)d_in[30], *bfv     = (const float*)d_in[31];
    const float* Wcat_f  = (const float*)d_in[32], *bcat_f  = (const float*)d_in[33];

    // Workspace: q | k | v | s_msg | dsum (fp32) | bf16 weight fragments
    float* ws   = (float*)d_ws;
    float* qbuf = ws;
    float* kbuf = qbuf + (long)kNT * kH;
    float* vbuf = kbuf + (long)kNS * kH;
    float* smsg = vbuf + (long)kNS * kH;
    float* dsum = smsg + (long)kNT * kH;
    u16* frag   = (u16*)(dsum + (long)kNT * 2 * kH);
    u16* fWs2     = frag;                    // 64x128
    u16* fWt2     = fWs2     + 64 * 128;     // 64x128
    u16* fWq      = fWt2     + 64 * 128;     // 128x128
    u16* fWk      = fWq      + 128 * 128;
    u16* fWv      = fWk      + 128 * 128;
    u16* fWdk     = fWv      + 128 * 128;
    u16* fWdv     = fWdk     + 128 * 128;
    u16* fWo_att  = fWdv     + 128 * 128;
    u16* fWs_msg  = fWo_att  + 128 * 128;    // 128x256
    u16* fWo_msg  = fWs_msg  + 128 * 256;    // 128x256
    u16* fWf      = fWo_msg  + 128 * 256;    // 128x384
    u16* fWcat_f  = fWf      + 128 * 384;    // 256x128
    u16* fWcat_msg= fWcat_f  + 256 * 128;    // 256x128
    u16* fWfinal  = fWcat_msg+ 256 * 128;    // 256x128

    float* ht  = (float*)d_out;            // (NT, 128)
    float* fji = ht + (long)kNT * kH;      // (E, 128)

    const int* src = ei;           // edge_index row 0
    const int* tgt = ei + kE;      // edge_index row 1

    // One-time weight pre-swizzle to bf16 WMMA-B fragments (stays hot in L2).
    prep_weight_kernel<<<64, 256, 0, stream>>>(Ws2, 64, 128, fWs2);
    prep_weight_kernel<<<64, 256, 0, stream>>>(Wt2, 64, 128, fWt2);
    prep_weight_kernel<<<64, 256, 0, stream>>>(Wq, 128, 128, fWq);
    prep_weight_kernel<<<64, 256, 0, stream>>>(Wk, 128, 128, fWk);
    prep_weight_kernel<<<64, 256, 0, stream>>>(Wv, 128, 128, fWv);
    prep_weight_kernel<<<64, 256, 0, stream>>>(Wdk, 128, 128, fWdk);
    prep_weight_kernel<<<64, 256, 0, stream>>>(Wdv, 128, 128, fWdv);
    prep_weight_kernel<<<64, 256, 0, stream>>>(Wo_att, 128, 128, fWo_att);
    prep_weight_kernel<<<64, 256, 0, stream>>>(Ws_msg, 128, 256, fWs_msg);
    prep_weight_kernel<<<64, 256, 0, stream>>>(Wo_msg, 128, 256, fWo_msg);
    prep_weight_kernel<<<64, 256, 0, stream>>>(Wf, 128, 384, fWf);
    prep_weight_kernel<<<64, 256, 0, stream>>>(Wcat_f, 256, 128, fWcat_f);
    prep_weight_kernel<<<64, 256, 0, stream>>>(Wcat_msg, 256, 128, fWcat_msg);
    prep_weight_kernel<<<64, 256, 0, stream>>>(Wfinal, 256, 128, fWfinal);

    // smsg and dsum are contiguous: one zeroing pass
    zero_kernel<<<1024, 256, 0, stream>>>(smsg, (long)kNT * kH + (long)kNT * 2 * kH);

    qkv_kernel<<<kNT / 16, 256, 0, stream>>>(x_s, x_t, fWq, bq, fWk, bk, fWv, bv,
                                             qbuf, kbuf, vbuf);

    edge_kernel<<<(int)(kE / 32), 256, 0, stream>>>(
        edge_attr, src, tgt, sphe, tors, dist,
        Ws1, fWs2, Wt1, fWt2, fWdk, bdk, fWdv, bdv, fWo_att, bo_att,
        fWs_msg, bs_msg, fWf, bfv, fWcat_f, bcat_f,
        qbuf, kbuf, vbuf, smsg, dsum, fji);

    final_kernel<<<kNT / 16, 256, 0, stream>>>(x_t, smsg, dsum, fWcat_msg,
                                               fWo_msg, bo_msg, fWfinal, bfinal, ht);
}
